// GroupedQueryAttention_53618371723503
// MI455X (gfx1250) — matrix operand order repaired and checked by reading, not compiled, not run on hardware
//
#include <hip/hip_runtime.h>
#include <hip/hip_bf16.h>
#include <math.h>

// ---------------- problem constants ----------------
static constexpr int NH   = 16;      // query heads
static constexpr int NKV  = 8;       // kv heads
static constexpr int HD   = 256;     // head dim
static constexpr int EMB  = 3584;
static constexpr int SEQ  = 2048;
static constexpr int NB   = 2;
static constexpr int NQKV = NH*HD + 2*NKV*HD;   // 8192
static constexpr int WIN  = 1024;
static constexpr float CAP = 50.0f;

// ---------------- vector types ----------------
typedef __attribute__((ext_vector_type(16))) __bf16 v16bf;
typedef __attribute__((ext_vector_type(8)))  __bf16 v8bf;
typedef __attribute__((ext_vector_type(4)))  __bf16 v4bf;
typedef __attribute__((ext_vector_type(8)))  float  v8f;
typedef __attribute__((ext_vector_type(4)))  float  v4f;

static __device__ __forceinline__ v8f vzero8() {
  v8f z;
#pragma unroll
  for (int i = 0; i < 8; ++i) z[i] = 0.0f;
  return z;
}
static __device__ __forceinline__ v16bf frag2(v8bf lo, v8bf hi) {
  union { v16bf v; v8bf h[2]; } u;
  u.h[0] = lo; u.h[1] = hi;
  return u.v;
}

// -------- CDNA5 async global->LDS copy (ASYNCcnt path, §15.18.3 op 98) ------
// Each lane copies 16B straight into LDS; low 32 bits of a generic pointer to
// a __shared__ object are the LDS byte address (flat-LDS maps addr[31:0]).
static __device__ __forceinline__ void async_copy_b128(void* lds_dst,
                                                       const void* gsrc) {
  unsigned lds = (unsigned)(size_t)lds_dst;
  asm volatile("global_load_async_to_lds_b128 %0, %1, off"
               :: "v"(lds), "v"(gsrc)
               : "memory");
}
static __device__ __forceinline__ void wait_async0() {
  asm volatile("s_wait_asynccnt 0x0" ::: "memory");
}

// A fragment: 16(M) x 32(K) bf16, row-major LDS tile, stride lda elements.
// Lane L<16 holds row M=L, K = {0..7,16..23}; lane L+16 holds K = {8..15,24..31}.
static __device__ __forceinline__ v16bf load_A_frag(const __bf16* p, int lda) {
  const int lane = threadIdx.x & 31;
  const int row  = lane & 15;
  const int half = lane >> 4;
  const __bf16* q = p + row * lda + half * 8;
  return frag2(*(const v8bf*)q, *(const v8bf*)(q + 16));
}
// B fragment: 32(K) x 16(N) bf16 from an N-major LDS tile (Bt[n][k], stride ldb).
static __device__ __forceinline__ v16bf load_B_frag(const __bf16* p, int ldb) {
  const int lane = threadIdx.x & 31;
  const int col  = lane & 15;
  const int half = lane >> 4;
  const __bf16* q = p + col * ldb + half * 8;
  return frag2(*(const v8bf*)q, *(const v8bf*)(q + 16));
}
static __device__ __forceinline__ v8f wmma_bf16(v16bf a, v16bf b, v8f c) {
  return __builtin_amdgcn_wmma_f32_16x16x32_bf16(false, a, false, b, (short)0, c,
                                                 false, false);
}

// ---------------- cast kernels ----------------
__global__ void cast_f32_bf16(const float* __restrict__ src,
                              __bf16* __restrict__ dst, long n) {
  long i = ((long)blockIdx.x * blockDim.x + threadIdx.x) * 4;
  if (i + 3 < n) {
    v4f v = *(const v4f*)(src + i);
    v4bf o;
    o[0] = (__bf16)v.x; o[1] = (__bf16)v.y; o[2] = (__bf16)v.z; o[3] = (__bf16)v.w;
    *(v4bf*)(dst + i) = o;
  }
}
// cast fp32 [rows x cols] into bf16 dst with row stride dst_stride at col offset.
__global__ void cast2d_f32_bf16(const float* __restrict__ src,
                                __bf16* __restrict__ dst, int rows, int cols,
                                int dst_stride, int dst_col0) {
  long idx = ((long)blockIdx.x * blockDim.x + threadIdx.x) * 4;
  long total = (long)rows * cols;
  if (idx >= total) return;
  int r = (int)(idx / cols);
  int c = (int)(idx % cols);        // cols % 4 == 0 for all our weights
  v4f v = *(const v4f*)(src + (long)r * cols + c);
  v4bf o;
  o[0] = (__bf16)v.x; o[1] = (__bf16)v.y; o[2] = (__bf16)v.z; o[3] = (__bf16)v.w;
  *(v4bf*)(dst + (long)r * dst_stride + dst_col0 + c) = o;
}

// ---------------- tiled WMMA GEMM: C[M,N] = A[M,K] * Bw[K,N] ----------------
// Block tile 64(M) x 256(N), K-step 32. 256 threads = 8 waves, each wave does
// a 32x64 sub-tile as 2x4 v_wmma_f32_16x16x32_bf16 accumulators.
// A tile is staged with global_load_async_to_lds_b128 (ASYNCcnt); B tile is
// transposed through VGPRs so B-fragment LDS reads stay contiguous.
template <typename OutT>
__global__ __launch_bounds__(256) void gemm_bf16_kernel(
    const __bf16* __restrict__ A, const __bf16* __restrict__ Bw,
    OutT* __restrict__ C, int K, int N) {
  __shared__ __bf16 As[64 * 40];    // 64 x 32 (+pad), row-major
  __shared__ __bf16 Bt[256 * 40];   // 256(n) x 32(k) (+pad), N-major
  const int t    = threadIdx.x;
  const int wave = t >> 5;
  const int lane = t & 31;
  const int wm   = wave & 1;        // 0..1  (M sub-tile)
  const int wn   = wave >> 1;       // 0..3  (N sub-tile)
  const long M0  = (long)blockIdx.y * 64;
  const long N0  = (long)blockIdx.x * 256;

  v8f acc[2][4];
#pragma unroll
  for (int i = 0; i < 2; ++i)
#pragma unroll
    for (int j = 0; j < 4; ++j) acc[i][j] = vzero8();

  const int arow = t >> 2;          // A tile: 64x32, 8 elems (16B)/thread
  const int acol = (t & 3) * 8;
  const int kb4  = (t >> 5) * 4;    // B tile: 32x256, 32 elems/thread
  const int nb8  = (t & 31) * 8;

  for (int k0 = 0; k0 < K; k0 += 32) {
    // stage A row-major: straight copy -> async global->LDS
    async_copy_b128(&As[arow * 40 + acol],
                    &A[(M0 + arow) * (long)K + k0 + acol]);
    // prefetch next K-step's slice of B into L2
    if (k0 + 32 < K)
      __builtin_prefetch(&Bw[(long)(k0 + 32 + kb4) * N + N0 + nb8], 0, 1);
    // stage B transposed (N-major) so fragment loads are contiguous
#pragma unroll
    for (int kk = 0; kk < 4; ++kk) {
      v8bf w = *(const v8bf*)&Bw[(long)(k0 + kb4 + kk) * N + N0 + nb8];
#pragma unroll
      for (int e = 0; e < 8; ++e) Bt[(nb8 + e) * 40 + kb4 + kk] = w[e];
    }
    wait_async0();
    __syncthreads();

    v16bf af[2], bfr[4];
#pragma unroll
    for (int i = 0; i < 2; ++i)
      af[i] = load_A_frag(&As[(wm * 32 + i * 16) * 40], 40);
#pragma unroll
    for (int j = 0; j < 4; ++j)
      bfr[j] = load_B_frag(&Bt[(wn * 64 + j * 16) * 40], 40);
#pragma unroll
    for (int i = 0; i < 2; ++i)
#pragma unroll
      for (int j = 0; j < 4; ++j) acc[i][j] = wmma_bf16(af[i], bfr[j], acc[i][j]);
    __syncthreads();
  }

  // epilogue: C layout lane=col(n), VGPR r -> row r + 8*half
  const int col = lane & 15, half = lane >> 4;
#pragma unroll
  for (int i = 0; i < 2; ++i)
#pragma unroll
    for (int j = 0; j < 4; ++j)
#pragma unroll
      for (int r = 0; r < 8; ++r) {
        long m = M0 + wm * 32 + i * 16 + r + half * 8;
        long n = N0 + wn * 64 + j * 16 + col;
        C[m * N + n] = (OutT)acc[i][j][r];
      }
}

// ---------------- RoPE on Q and K (in-place on bf16 qkv) ----------------
__global__ void rope_kernel(__bf16* __restrict__ qkv) {
  const long row  = blockIdx.x;            // 0 .. NB*SEQ-1
  const int  head = blockIdx.y;            // 0 .. NH+NKV-1
  const int  s    = (int)(row % SEQ);
  const int  d    = threadIdx.x;           // 0..127  (pair d, d+128)
  __bf16* p = qkv + row * NQKV +
              (head < NH ? head * HD : NH * HD + (head - NH) * HD);
  float inv = __powf(10000.0f, -(float)d / 128.0f);   // 10000^(-2d/256)
  float ang = (float)s * inv;
  float c = __cosf(ang), sn = __sinf(ang);
  float x1 = (float)p[d], x2 = (float)p[d + 128];
  p[d]       = (__bf16)(x1 * c - x2 * sn);
  p[d + 128] = (__bf16)(x2 * c + x1 * sn);
}

// ---------------- flash attention with soft-cap + sliding window ------------
// grid: (SEQ/64, NH, NB), block 128 (4 waves). Wave w owns q rows q0+16w..+15.
static constexpr int LQK = 264;  // Q/K LDS row stride (bf16), 528B = 33*16
static constexpr int LV  = 72;   // Vt / P LDS row stride, 144B = 9*16
static constexpr int SMEM_ATTN =
    (64 * LQK + 64 * LQK + 256 * LV + 4 * 16 * LV) * 2;   // 113,664 B

__global__ __launch_bounds__(128) void attn_kernel(
    const __bf16* __restrict__ qkv, __bf16* __restrict__ ao) {
  extern __shared__ __align__(16) char smem_raw[];
  __bf16* Qs = (__bf16*)smem_raw;        // 64 x LQK  (row-major, 16 q rows/wave)
  __bf16* Ks = Qs + 64 * LQK;            // 64 x LQK  (row-major == N-major for QK^T)
  __bf16* Vt = Ks + 64 * LQK;            // 256(d) x LV (V transposed: N-major for PV)
  __bf16* Ps = Vt + 256 * LV;            // 4 waves x 16 x LV (P staging, C->A relayout)

  const int t = threadIdx.x;
  const int wave = t >> 5, lane = t & 31;
  const int col = lane & 15, half = lane >> 4;
  const int q0 = blockIdx.x * 64;
  const int h  = blockIdx.y;
  const int b  = blockIdx.z;
  const long rowbase = (long)b * SEQ;
  const int kofs = NH * HD + (h >> 1) * HD;
  const int vofs = NH * HD + NKV * HD + (h >> 1) * HD;
  const float scale = 0.0625f;           // 1/sqrt(256)

  // load Q tile 64 x 256 via async global->LDS copies
  for (int c8 = t; c8 < 64 * 32; c8 += 128) {
    int r = c8 >> 5, cg = c8 & 31;
    async_copy_b128(&Qs[r * LQK + cg * 8],
                    &qkv[(rowbase + q0 + r) * NQKV + h * HD + cg * 8]);
  }
  wait_async0();
  __syncthreads();

  // hoist Q fragments (8 K-chunks covering D=256)
  v16bf qf[8];
#pragma unroll
  for (int kc = 0; kc < 8; ++kc)
    qf[kc] = load_A_frag(&Qs[(wave * 16) * LQK + kc * 32], LQK);

  v8f O[16];
#pragma unroll
  for (int d0 = 0; d0 < 16; ++d0) O[d0] = vzero8();
  float mrow[8], lrow[8];
#pragma unroll
  for (int r = 0; r < 8; ++r) { mrow[r] = -INFINITY; lrow[r] = 0.0f; }

  const int kstart = (q0 - WIN) > 0 ? (q0 - WIN) : 0;       // 64-aligned
  const int kend   = (q0 + 64 + WIN) < SEQ ? (q0 + 64 + WIN) : SEQ;
  __bf16* PsW = Ps + wave * 16 * LV;

  for (int kb = kstart; kb < kend; kb += 64) {
    // stage K: async straight copy; stage V transposed through VGPRs
    for (int c8 = t; c8 < 64 * 32; c8 += 128) {
      int r = c8 >> 5, cg = c8 & 31;
      long grow = (rowbase + kb + r) * NQKV;
      async_copy_b128(&Ks[r * LQK + cg * 8], &qkv[grow + kofs + cg * 8]);
      v8bf vv = *(const v8bf*)&qkv[grow + vofs + cg * 8];
#pragma unroll
      for (int e = 0; e < 8; ++e) Vt[(cg * 8 + e) * LV + r] = vv[e];
    }
    wait_async0();
    __syncthreads();

    // scores: S[16x64] = Q(16x256) @ K^T(256x64)
    v8f sc[4];
#pragma unroll
    for (int j = 0; j < 4; ++j) {
      sc[j] = vzero8();
#pragma unroll
      for (int kc = 0; kc < 8; ++kc) {
        v16bf kf = load_B_frag(&Ks[(j * 16) * LQK + kc * 32], LQK);
        sc[j] = wmma_bf16(qf[kc], kf, sc[j]);
      }
    }

    // scale, tanh soft-cap, window mask; per-row block max
    float bmax[8];
#pragma unroll
    for (int r = 0; r < 8; ++r) bmax[r] = -INFINITY;
#pragma unroll
    for (int j = 0; j < 4; ++j) {
      int kpos = kb + j * 16 + col;
#pragma unroll
      for (int r = 0; r < 8; ++r) {
        int qpos = q0 + wave * 16 + r + half * 8;
        float sv = CAP * tanhf(sc[j][r] * (scale / CAP));
        int dlt = qpos - kpos; dlt = dlt < 0 ? -dlt : dlt;
        sv = (dlt <= WIN) ? sv : -INFINITY;
        sc[j][r] = sv;
        bmax[r] = fmaxf(bmax[r], sv);
      }
    }

    // online softmax stats (reduce within each 16-lane half)
    float corr[8];
#pragma unroll
    for (int r = 0; r < 8; ++r) {
      float v = bmax[r];
      v = fmaxf(v, __shfl_xor(v, 1, 32));
      v = fmaxf(v, __shfl_xor(v, 2, 32));
      v = fmaxf(v, __shfl_xor(v, 4, 32));
      v = fmaxf(v, __shfl_xor(v, 8, 32));
      float mnew = fmaxf(mrow[r], v);
      corr[r] = __expf(mrow[r] - mnew);
      mrow[r] = mnew;
      lrow[r] *= corr[r];
    }

    // p = exp(s - m): row-sum + stage into LDS in A layout source form
    float rsum[8];
#pragma unroll
    for (int r = 0; r < 8; ++r) rsum[r] = 0.0f;
#pragma unroll
    for (int j = 0; j < 4; ++j)
#pragma unroll
      for (int r = 0; r < 8; ++r) {
        float p = __expf(sc[j][r] - mrow[r]);
        rsum[r] += p;
        PsW[(r + half * 8) * LV + j * 16 + col] = (__bf16)p;
      }
#pragma unroll
    for (int r = 0; r < 8; ++r) {
      float v = rsum[r];
      v += __shfl_xor(v, 1, 32);
      v += __shfl_xor(v, 2, 32);
      v += __shfl_xor(v, 4, 32);
      v += __shfl_xor(v, 8, 32);
      lrow[r] += v;
    }

    // rescale O, then O += P(16x64) @ V(64x256)
#pragma unroll
    for (int d0 = 0; d0 < 16; ++d0)
#pragma unroll
      for (int r = 0; r < 8; ++r) O[d0][r] *= corr[r];

    v16bf pf0 = load_A_frag(PsW, LV);          // intra-wave LDS RAW (in order)
    v16bf pf1 = load_A_frag(PsW + 32, LV);
#pragma unroll
    for (int d0 = 0; d0 < 16; ++d0) {
      v16bf vf0 = load_B_frag(&Vt[(d0 * 16) * LV], LV);
      v16bf vf1 = load_B_frag(&Vt[(d0 * 16) * LV + 32], LV);
      O[d0] = wmma_bf16(pf0, vf0, O[d0]);
      O[d0] = wmma_bf16(pf1, vf1, O[d0]);
    }
    __syncthreads();   // Ks/Vt reused next k-block
  }

  // normalize and store (bf16, feeds the final WMMA GEMM)
#pragma unroll
  for (int r = 0; r < 8; ++r) lrow[r] = 1.0f / lrow[r];
#pragma unroll
  for (int d0 = 0; d0 < 16; ++d0)
#pragma unroll
    for (int r = 0; r < 8; ++r) {
      long m = rowbase + q0 + wave * 16 + r + half * 8;
      ao[m * (NH * HD) + h * HD + d0 * 16 + col] = (__bf16)(O[d0][r] * lrow[r]);
    }
}

// ---------------- host side ----------------
extern "C" void kernel_launch(void* const* d_in, const int* in_sizes, int n_in,
                              void* d_out, int out_size, void* d_ws,
                              size_t ws_size, hipStream_t stream) {
  const float* hs = (const float*)d_in[0];
  const float* Wq = (const float*)d_in[1];
  const float* Wk = (const float*)d_in[2];
  const float* Wv = (const float*)d_in[3];
  const float* Wo = (const float*)d_in[4];

  char* ws = (char*)d_ws;
  // ws layout (bytes), all 256-aligned:
  __bf16* hsB   = (__bf16*)(ws + 0);            //  2*2048*3584  = 29,360,128 B
  __bf16* wqkvB = (__bf16*)(ws + 29360128);     //  3584*8192*2  = 58,720,256 B
  __bf16* woB   = (__bf16*)(ws + 88080384);     //  4096*3584*2  = 29,360,128 B
  __bf16* qkvB  = (__bf16*)(ws + 117440512);    //  4096*8192*2  = 67,108,864 B
  __bf16* aoB   = (__bf16*)(ws + 184549376);    //  4096*4096*2  = 33,554,432 B

  const long M = (long)NB * SEQ;                // 4096 token rows

  // 1) casts
  long nhs = M * EMB;
  cast_f32_bf16<<<(int)((nhs / 4 + 255) / 256), 256, 0, stream>>>(hs, hsB, nhs);
  cast2d_f32_bf16<<<(int)(((long)EMB * NH * HD / 4 + 255) / 256), 256, 0, stream>>>(
      Wq, wqkvB, EMB, NH * HD, NQKV, 0);
  cast2d_f32_bf16<<<(int)(((long)EMB * NKV * HD / 4 + 255) / 256), 256, 0, stream>>>(
      Wk, wqkvB, EMB, NKV * HD, NQKV, NH * HD);
  cast2d_f32_bf16<<<(int)(((long)EMB * NKV * HD / 4 + 255) / 256), 256, 0, stream>>>(
      Wv, wqkvB, EMB, NKV * HD, NQKV, NH * HD + NKV * HD);
  long nwo = (long)NH * HD * EMB;
  cast_f32_bf16<<<(int)((nwo / 4 + 255) / 256), 256, 0, stream>>>(Wo, woB, nwo);

  // 2) QKV projection: [4096,3584] @ [3584,8192] -> bf16
  gemm_bf16_kernel<__bf16><<<dim3(NQKV / 256, (int)(M / 64)), 256, 0, stream>>>(
      hsB, wqkvB, qkvB, EMB, NQKV);

  // 3) RoPE on Q and K
  rope_kernel<<<dim3((int)M, NH + NKV), 128, 0, stream>>>(qkvB);

  // 4) windowed soft-cap flash attention
  attn_kernel<<<dim3(SEQ / 64, NH, NB), 128, SMEM_ATTN, stream>>>(qkvB, aoB);

  // 5) output projection: [4096,4096] @ [4096,3584] -> fp32 d_out
  gemm_bf16_kernel<float><<<dim3(EMB / 256, (int)(M / 64)), 256, 0, stream>>>(
      aoB, woB, (float*)d_out, NH * HD, EMB);
}
